// RPN1D_9869834846298
// MI455X (gfx1250) — matrix operand order repaired and checked by reading, not compiled, not run on hardware
//
#include <hip/hip_runtime.h>

typedef __attribute__((ext_vector_type(16))) _Float16 v16h;
typedef __attribute__((ext_vector_type(8)))  float    v8f;

#define BDIM  8
#define CDIM  512
#define LF    16384
#define ADIM  7
#define KTOT  1536            // 3 * 512 (im2col K)
#define NKC   48              // KTOT / 32
#define LT    64              // L positions per workgroup
#define NROW  (LT + 2)        // +halo for 3-tap conv
#define LDST  520             // padded LDS row stride in halves (512 + 8)
#define OBJ_SZ   (BDIM * LF * ADIM)        // 917504
#define REG_SZ   (BDIM * LF * ADIM * 2)    // 1835008
#define ANCH_OFF (OBJ_SZ + REG_SZ)         // 2752512

// 16-bit A-matrix half-index -> K-within-chunk mapping (contiguous, matches B):
// lanes 0-15: halves 0..15 -> K 0..15
// lanes 16-31: halves 0..15 -> K 16..31
__device__ __forceinline__ int kmap(int h, int lane) {
    return h + ((lane < 16) ? 0 : 16);
}

// ---- pack conv weights (fp32 (O,I,3)) into f16 A-fragment layout in ws ----
// ws layout: frag(mtile 0..31, kc 0..47) -> 32 lanes x 16 halves contiguous.
__global__ void pack_conv_w(const float* __restrict__ conv_w,
                            _Float16* __restrict__ wsA) {
    int t   = blockIdx.x * 256 + threadIdx.x;   // 32*48*512 = 786432 exactly
    int mt  = t / (NKC * 512);
    int rem = t % (NKC * 512);
    int kc  = rem >> 9;
    int r2  = rem & 511;
    int lane = r2 >> 4;
    int h    = r2 & 15;
    int m   = mt * 16 + (lane & 15);
    int K   = kc * 32 + kmap(h, lane);          // K index = tap*512 + c_in
    int tap = K >> 9;
    int ci  = K & 511;
    wsA[t] = (_Float16)conv_w[(m * 512 + ci) * 3 + tap];
}

// ---- pack head weights (obj 7xC, reg 14xC, padded to 32 rows) ----
__global__ void pack_head_w(const float* __restrict__ obj_w,
                            const float* __restrict__ reg_w,
                            _Float16* __restrict__ wsH) {
    int t   = blockIdx.x * 256 + threadIdx.x;   // 2*16*512 = 16384 exactly
    int mt  = t >> 13;
    int rem = t & 8191;
    int kc  = rem >> 9;
    int r2  = rem & 511;
    int lane = r2 >> 4;
    int h    = r2 & 15;
    int m = mt * 16 + (lane & 15);
    int c = kc * 32 + kmap(h, lane);
    float v = 0.f;
    if (m < ADIM)          v = obj_w[m * CDIM + c];
    else if (m < 3 * ADIM) v = reg_w[(m - ADIM) * CDIM + c];
    wsH[t] = (_Float16)v;
}

// ---- anchors ----
__global__ void anchors_k(float* __restrict__ out) {
    int t = blockIdx.x * 256 + threadIdx.x;     // LF*7 = 114688 = 448*256
    const float LEN[ADIM] = {1.f, 2.f, 3.f, 4.f, 5.f, 7.f, 9.f};
    int l = t / ADIM, a = t % ADIM;
    float c = (float)l + 0.5f;
    float w = 0.5f * LEN[a];
    out[ANCH_OFF + 2 * t]     = c - w;
    out[ANCH_OFF + 2 * t + 1] = c + w;
}

// ---- fused conv(3) + ReLU + obj/reg heads, all WMMA f16 ----
__launch_bounds__(256, 1)
__global__ void rpn_fused(const float* __restrict__ feat,
                          const _Float16* __restrict__ wsA,
                          const _Float16* __restrict__ wsH,
                          const float* __restrict__ conv_b,
                          const float* __restrict__ obj_b,
                          const float* __restrict__ reg_b,
                          float* __restrict__ out) {
    __shared__ __align__(16) _Float16 smem[NROW * LDST];   // 68,640 B, reused
    const int tid  = threadIdx.x;
    const int lane = tid & 31;
    const int wave = tid >> 5;
    const int nlo  = lane & 15;
    const int hi16 = (lane >> 4) & 1;
    const int l0   = blockIdx.x * LT;
    const int b    = blockIdx.y;

    // Phase 1: feat tile -> LDS, transposed [l][c], f16, zero-padded halo.
    {
        const float* fb = feat + (size_t)b * CDIM * LF;
        for (int idx = tid; idx < CDIM * NROW; idx += 256) {
            int c = idx / NROW;                 // consecutive tid -> consecutive l
            int j = idx - c * NROW;             // 0..65 -> l = l0-1+j
            int l = l0 - 1 + j;
            float v = (l >= 0 && l < LF) ? fb[(size_t)c * LF + l] : 0.f;
            smem[j * LDST + c] = (_Float16)v;
        }
    }
    __syncthreads();

    // Phase 2: conv GEMM. Wave owns c_out rows [wave*64, wave*64+64).
    v8f acc[4][4];
    const v8f vzero = {0.f, 0.f, 0.f, 0.f, 0.f, 0.f, 0.f, 0.f};
#pragma unroll
    for (int mt = 0; mt < 4; ++mt)
#pragma unroll
        for (int nt = 0; nt < 4; ++nt) acc[mt][nt] = vzero;

    for (int kc = 0; kc < NKC; ++kc) {
        const int tap = kc >> 4;                // which conv tap
        const int c0  = (kc & 15) << 5;         // c_in base of this K-chunk
        v16h bf[4];
#pragma unroll
        for (int nt = 0; nt < 4; ++nt) {        // B frag: 2x b128 LDS loads
            const _Float16* p = &smem[(nt * 16 + nlo + tap) * LDST + c0 + hi16 * 16];
            ((float4*)&bf[nt])[0] = ((const float4*)p)[0];
            ((float4*)&bf[nt])[1] = ((const float4*)p)[1];
        }
#pragma unroll
        for (int mt = 0; mt < 4; ++mt) {
            const int mtile = wave * 4 + mt;
            const _Float16* pa = wsA + (((size_t)(mtile * NKC + kc)) * 32 + lane) * 16;
            v16h af;
            ((float4*)&af)[0] = ((const float4*)pa)[0];
            ((float4*)&af)[1] = ((const float4*)pa)[1];
#pragma unroll
            for (int nt = 0; nt < 4; ++nt)
                acc[mt][nt] = __builtin_amdgcn_wmma_f32_16x16x32_f16(
                    false, af, false, bf[nt], (short)0, acc[mt][nt], false, false);
        }
    }
    __syncthreads();                            // everyone done reading feat

    // Phase 3: bias + ReLU, write h transposed [n][c_out] over the same LDS.
#pragma unroll
    for (int mt = 0; mt < 4; ++mt) {
        const int mbase = (wave * 4 + mt) * 16 + hi16 * 8;
#pragma unroll
        for (int nt = 0; nt < 4; ++nt) {
            const int n = nt * 16 + nlo;
            _Float16 tmp[8] __attribute__((aligned(16)));
#pragma unroll
            for (int r = 0; r < 8; ++r) {
                float v = acc[mt][nt][r] + conv_b[mbase + r];
                tmp[r] = (_Float16)(v > 0.f ? v : 0.f);
            }
            *(float4*)&smem[n * LDST + mbase] = *(const float4*)tmp;
        }
    }
    __syncthreads();

    // Phase 4: head GEMM. One 16x16 output tile per wave (2 row x 4 col tiles).
    const int mt2 = wave & 1;
    const int nt2 = wave >> 1;
    v8f acc2 = vzero;
    for (int kc = 0; kc < 16; ++kc) {
        const _Float16* pa = wsH + (((size_t)(mt2 * 16 + kc)) * 32 + lane) * 16;
        v16h af;
        ((float4*)&af)[0] = ((const float4*)pa)[0];
        ((float4*)&af)[1] = ((const float4*)pa)[1];
        const _Float16* pb = &smem[(nt2 * 16 + nlo) * LDST + kc * 32 + hi16 * 16];
        v16h bf;
        ((float4*)&bf)[0] = ((const float4*)pb)[0];
        ((float4*)&bf)[1] = ((const float4*)pb)[1];
        acc2 = __builtin_amdgcn_wmma_f32_16x16x32_f16(
            false, af, false, bf, (short)0, acc2, false, false);
    }

    // Epilogue: scatter obj / reg with biases.
    const int n = nt2 * 16 + nlo;
    const int l = l0 + n;
    const int mbase = mt2 * 16 + hi16 * 8;
#pragma unroll
    for (int r = 0; r < 8; ++r) {
        const int m = mbase + r;
        if (m < ADIM) {
            out[((size_t)b * LF + l) * ADIM + m] = acc2[r] + obj_b[m];
        } else if (m < 3 * ADIM) {
            // reg[b,l,o] with o = m-7; flat offset OBJ_SZ + (b*LF+l)*14 + o
            out[OBJ_SZ + ((size_t)b * LF + l) * (2 * ADIM) + (m - ADIM)] =
                acc2[r] + reg_b[m - ADIM];
        }
    }
}

extern "C" void kernel_launch(void* const* d_in, const int* in_sizes, int n_in,
                              void* d_out, int out_size, void* d_ws, size_t ws_size,
                              hipStream_t stream) {
    (void)in_sizes; (void)n_in; (void)out_size; (void)ws_size;
    const float* feat   = (const float*)d_in[0];
    const float* conv_w = (const float*)d_in[1];
    const float* conv_b = (const float*)d_in[2];
    const float* obj_w  = (const float*)d_in[3];
    const float* obj_b  = (const float*)d_in[4];
    const float* reg_w  = (const float*)d_in[5];
    const float* reg_b  = (const float*)d_in[6];
    float* out = (float*)d_out;

    _Float16* wsA = (_Float16*)d_ws;            // 32*48*512 halves = 1.5 MB
    _Float16* wsH = wsA + 32 * NKC * 512;       // 2*16*512 halves  = 32 KB

    pack_conv_w<<<3072, 256, 0, stream>>>(conv_w, wsA);
    pack_head_w<<<64,   256, 0, stream>>>(obj_w, reg_w, wsH);
    anchors_k  <<<448,  256, 0, stream>>>(out);

    dim3 grid(LF / LT, BDIM);                   // (256, 8)
    rpn_fused<<<grid, 256, 0, stream>>>(feat, wsA, wsH, conv_b, obj_b, reg_b, out);
}
